// MultiHeadDeformableAttention_37014028157236
// MI455X (gfx1250) — matrix operand order repaired and checked by reading, not compiled, not run on hardware
//
#include <hip/hip_runtime.h>
#include <hip/hip_bf16.h>

typedef __attribute__((ext_vector_type(16))) _Float16 v16h;
typedef __attribute__((ext_vector_type(8)))  _Float16 v8h;
typedef __attribute__((ext_vector_type(8)))  float    v8f;
typedef __attribute__((ext_vector_type(2)))  float    v2f;

#define Bq    2
#define Cc    128
#define Hh    48
#define Ww    48
#define HEADS 4
#define Gg    4
#define DHd   32
#define DFs   4
#define OKS   6
#define PADc  1
#define HD    12
#define WD    12
#define HW    2304   // 48*48
#define Jj    144    // 12*12
#define JP    160    // padded j (5*32)
#define BG    8      // B*G == B*HEADS
#define CPB   32

// ---------------------------------------------------------------------------
// 1) grouped 1x1 conv -> q  (fp32 for offset path, f16*scale transposed for WMMA A)
// ---------------------------------------------------------------------------
__global__ void q_kernel(const float* __restrict__ x, const float* __restrict__ wq,
                         float* __restrict__ qf, _Float16* __restrict__ qT) {
  __shared__ float wqs[32 * 32];
  int bg = blockIdx.x;
  int b = bg / Gg, g = bg % Gg;
  int tid = threadIdx.x;
  for (int t = tid; t < 1024; t += 256) wqs[t] = wq[g * 1024 + t];
  __syncthreads();
  int hw = blockIdx.y * 256 + tid;
  float xl[32];
#pragma unroll
  for (int c = 0; c < 32; ++c)
    xl[c] = x[((size_t)b * Cc + g * 32 + c) * HW + hw];
  const float scale = 0.17677669529663687f; // 32^-0.5
  for (int o = 0; o < 32; ++o) {
    float acc = 0.f;
#pragma unroll
    for (int c = 0; c < 32; ++c) acc = fmaf(wqs[o * 32 + c], xl[c], acc);
    qf[((size_t)bg * 32 + o) * HW + hw] = acc;
    qT[((size_t)bg * HW + hw) * 32 + o] = (_Float16)(acc * scale);
  }
}

// ---------------------------------------------------------------------------
// 2) depthwise conv k=6 s=4 p=1 -> GELU -> 1x1 proj -> tanh*4 -> grids
// ---------------------------------------------------------------------------
__global__ void offsets_kernel(const float* __restrict__ q, const float* __restrict__ wdw,
                               const float* __restrict__ bdw, const float* __restrict__ wproj,
                               float* __restrict__ vn, float* __restrict__ pxy) {
  int bg = blockIdx.x, p = blockIdx.y;
  int oy = p / WD, ox = p % WD;
  int c = threadIdx.x;
  const float* qc = q + ((size_t)bg * 32 + c) * HW;
  float acc = 0.f;
  for (int ky = 0; ky < OKS; ++ky) {
    int iy = oy * DFs - PADc + ky;
    if (iy < 0 || iy >= Hh) continue;
    for (int kx = 0; kx < OKS; ++kx) {
      int ix = ox * DFs - PADc + kx;
      if (ix < 0 || ix >= Ww) continue;
      acc += qc[iy * Ww + ix] * wdw[c * 36 + ky * OKS + kx];
    }
  }
  acc += bdw[c];
  float gel = 0.5f * acc * (1.f + erff(acc * 0.70710678118654752f));
  __shared__ float sh[32];
  __shared__ float off2[2];
  sh[c] = gel;
  __syncthreads();
  if (c < 2) {
    float s = 0.f;
    for (int cc = 0; cc < 32; ++cc) s += wproj[c * 32 + cc] * sh[cc];
    off2[c] = tanhf(s) * 4.0f;
  }
  __syncthreads();
  if (c == 0) {
    float vg0 = (float)ox + off2[0];
    float vg1 = (float)oy + off2[1];
    float vn0 = 2.f * vg0 / (float)(HD - 1) - 1.f;
    float vn1 = 2.f * vg1 / (float)(WD - 1) - 1.f;
    float px = ((vn0 + 1.f) * (float)Ww - 1.f) * 0.5f;
    float py = ((vn1 + 1.f) * (float)Hh - 1.f) * 0.5f;
    vn[(bg * 2 + 0) * Jj + p] = vn0;
    vn[(bg * 2 + 1) * Jj + p] = vn1;
    pxy[(bg * 2 + 0) * Jj + p] = px;
    pxy[(bg * 2 + 1) * Jj + p] = py;
  }
}

// ---------------------------------------------------------------------------
// 3) bilinear sampling with zero padding (grid_sample semantics)
// ---------------------------------------------------------------------------
__global__ void sample_kernel(const float* __restrict__ x, const float* __restrict__ pxy,
                              float* __restrict__ kv) {
  int bg = blockIdx.x, p = blockIdx.y, c = threadIdx.x;
  float px = pxy[(bg * 2 + 0) * Jj + p];
  float py = pxy[(bg * 2 + 1) * Jj + p];
  int b = bg / Gg, g = bg % Gg;
  const float* img = x + ((size_t)b * Cc + g * 32 + c) * HW;
  float x0 = floorf(px), y0 = floorf(py);
  float wx = px - x0, wy = py - y0;
  float out = 0.f;
#pragma unroll
  for (int dy = 0; dy < 2; ++dy)
#pragma unroll
    for (int dx = 0; dx < 2; ++dx) {
      float xc = x0 + dx, yc = y0 + dy;
      float w = (dx ? wx : 1.f - wx) * (dy ? wy : 1.f - wy);
      bool valid = (xc >= 0.f) && (xc <= (float)(Ww - 1)) && (yc >= 0.f) && (yc <= (float)(Hh - 1));
      int xi = (int)fminf(fmaxf(xc, 0.f), (float)(Ww - 1));
      int yi = (int)fminf(fmaxf(yc, 0.f), (float)(Hh - 1));
      out += valid ? w * img[yi * Ww + xi] : 0.f;
    }
  kv[((size_t)bg * 32 + c) * Jj + p] = out;
}

// ---------------------------------------------------------------------------
// 4) k = wk@kv, v = wv@kv (small grouped GEMMs)
// ---------------------------------------------------------------------------
__global__ void kv_kernel(const float* __restrict__ kv, const float* __restrict__ wk,
                          const float* __restrict__ wv, float* __restrict__ kf,
                          float* __restrict__ vf) {
  int idx = blockIdx.x * blockDim.x + threadIdx.x;
  if (idx >= BG * 32 * Jj) return;
  int j = idx % Jj;
  int o = (idx / Jj) % 32;
  int bg = idx / (Jj * 32);
  int g = bg % Gg;
  float ak = 0.f, av = 0.f;
  for (int c = 0; c < 32; ++c) {
    float v = kv[((size_t)bg * 32 + c) * Jj + j];
    ak = fmaf(wk[g * 1024 + o * 32 + c], v, ak);
    av = fmaf(wv[g * 1024 + o * 32 + c], v, av);
  }
  kf[idx] = ak;
  vf[idx] = av;
}

// ---------------------------------------------------------------------------
// 5) pack operands into per-lane WMMA layouts (32B contiguous per lane)
// ---------------------------------------------------------------------------
__global__ void pack_k_kernel(const float* __restrict__ kf, _Float16* __restrict__ kB) {
  int idx = blockIdx.x * 32 + threadIdx.x; // (bh*10 + jt)*32 + lane
  int lane = idx & 31;
  int tile = idx >> 5;
  int jt = tile % 10, bh = tile / 10;
  int j = jt * 16 + (lane & 15);
  int kh = lane >> 4;
#pragma unroll
  for (int h = 0; h < 16; ++h) {
    int d = ((h < 8) ? 0 : 16) + kh * 8 + (h & 7);
    float v = (j < Jj) ? kf[((size_t)bh * 32 + d) * Jj + j] : 0.f;
    kB[(size_t)idx * 16 + h] = (_Float16)v;
  }
}

__global__ void pack_v_kernel(const float* __restrict__ vf, _Float16* __restrict__ vB) {
  int idx = blockIdx.x * 32 + threadIdx.x; // ((bh*5+kt)*2+nt)*32 + lane
  int lane = idx & 31;
  int tile = idx >> 5;
  int nt = tile & 1;
  int kt = (tile >> 1) % 5;
  int bh = tile / 10;
  int d = nt * 16 + (lane & 15);
  int kh = lane >> 4;
#pragma unroll
  for (int h = 0; h < 16; ++h) {
    int j = kt * 32 + ((h < 8) ? 0 : 16) + kh * 8 + (h & 7);
    float v = (j < Jj) ? vf[((size_t)bh * 32 + d) * Jj + j] : 0.f;
    vB[(size_t)idx * 16 + h] = (_Float16)v;
  }
}

// W1^T (CPB layer-2) in A-matrix layout: M = out channel, K = in channel
__global__ void prep_w1t_kernel(const float* __restrict__ w1, _Float16* __restrict__ w1a) {
  int tid = threadIdx.x; // 64 threads: mt*32 + lane
  int mt = tid >> 5, lane = tid & 31;
  int m = lane & 15;
  for (int h = 0; h < 16; ++h) {
    int k = ((lane >= 16) ? 16 : 0) + h;
    w1a[(size_t)tid * 16 + h] = (_Float16)w1[k * CPB + mt * 16 + m];
  }
}

// wo^T in f32 B-matrix (4x16) lane layout: per lane 2 consecutive K values
__global__ void prep_wo_kernel(const float* __restrict__ wo, float* __restrict__ woB) {
  int idx = blockIdx.x * blockDim.x + threadIdx.x; // nt(8) * kc(32) * lane(32)
  int lane = idx & 31;
  int kc = (idx >> 5) & 31;
  int nt = idx >> 10;
  int o = nt * 16 + (lane & 15);
  int kbase = kc * 4 + ((lane >= 16) ? 2 : 0);
  woB[(size_t)idx * 2 + 0] = wo[o * Cc + kbase + 0];
  woB[(size_t)idx * 2 + 1] = wo[o * Cc + kbase + 1];
}

// ---------------------------------------------------------------------------
// 6) CPB relative-position-bias MLP (dominant compute, ~6 GFLOP)
//    transposed orientation: D = W1^T (A, const) x h1^T (B, per-tile)
//    2 tiles / iteration: the second tile's Bh build + WMMAs fill the first
//    tile's WMMA->VALU hazard slots; layer-3 dot uses 2 accumulator chains.
// ---------------------------------------------------------------------------
__global__ void __launch_bounds__(32)
cpb_kernel(const float* __restrict__ vn, const _Float16* __restrict__ w1a,
           const float* __restrict__ w0, const float* __restrict__ b0,
           const float* __restrict__ b1, const float* __restrict__ w2,
           const float* __restrict__ b2v, float* __restrict__ bias) {
  int bg = blockIdx.x, i = blockIdx.y;
  int lane = threadIdx.x;
  int yq = i / Ww, xq = i % Ww;
  float qn0 = 2.f * (float)xq / (float)(Ww - 1) - 1.f;
  float qn1 = 2.f * (float)yq / (float)(Hh - 1) - 1.f;
  int chb = (lane >= 16) ? 16 : 0;
  float w00[16], w01[16], b0l[16];
#pragma unroll
  for (int h = 0; h < 16; ++h) {
    w00[h] = w0[chb + h];
    w01[h] = w0[32 + chb + h];
    b0l[h] = b0[chb + h];
  }
  v16h W1A0 = *(const v16h*)(w1a + (0 * 32 + lane) * 16);
  v16h W1A1 = *(const v16h*)(w1a + (1 * 32 + lane) * 16);
  int base = (lane >> 4) * 8;
  float b1l0[8], b1l1[8], w2l0[8], w2l1[8];
#pragma unroll
  for (int r = 0; r < 8; ++r) {
    b1l0[r] = b1[base + r];
    b1l1[r] = b1[16 + base + r];
    w2l0[r] = w2[base + r];
    w2l1[r] = w2[16 + base + r];
  }
  float b2s = b2v[0];
  int p = lane & 15;
  const float* vn0p = vn + (bg * 2 + 0) * Jj;
  const float* vn1p = vn + (bg * 2 + 1) * Jj;
  float* brow = bias + ((size_t)bg * HW + i) * Jj;

  auto build_B = [&](int j) -> v16h {
    float p0 = qn0 - vn0p[j];
    float p1 = qn1 - vn1p[j];
    float s0 = copysignf(log1pf(fabsf(p0)), p0);
    float s1 = copysignf(log1pf(fabsf(p1)), p1);
    v16h Bh;
#pragma unroll
    for (int h = 0; h < 16; ++h) {
      float hv = fmaf(s0, w00[h], fmaf(s1, w01[h], b0l[h]));
      Bh[h] = (_Float16)fmaxf(hv, 0.f);
    }
    return Bh;
  };
  auto dot3 = [&](const v8f& c0, const v8f& c1) -> float {
    float pa = 0.f, pb = 0.f; // two independent chains
#pragma unroll
    for (int r = 0; r < 8; ++r) {
      pa = fmaf(fmaxf(c0[r], 0.f), w2l0[r], pa);
      pb = fmaf(fmaxf(c1[r], 0.f), w2l1[r], pb);
    }
    return pa + pb;
  };

  for (int t = 0; t < 8; t += 2) { // pipelined pairs of j-tiles
    int ja = t * 16 + p;
    int jb = (t + 1) * 16 + p;
    v16h Ba = build_B(ja);
    v16h Bb = build_B(jb);
    v8f a0, a1, d0, d1;
#pragma unroll
    for (int r = 0; r < 8; ++r) {
      a0[r] = b1l0[r]; a1[r] = b1l1[r];
      d0[r] = b1l0[r]; d1[r] = b1l1[r];
    }
    a0 = __builtin_amdgcn_wmma_f32_16x16x32_f16(false, W1A0, false, Ba, (short)0, a0, false, false);
    a1 = __builtin_amdgcn_wmma_f32_16x16x32_f16(false, W1A1, false, Ba, (short)0, a1, false, false);
    d0 = __builtin_amdgcn_wmma_f32_16x16x32_f16(false, W1A0, false, Bb, (short)0, d0, false, false);
    d1 = __builtin_amdgcn_wmma_f32_16x16x32_f16(false, W1A1, false, Bb, (short)0, d1, false, false);
    float pa = dot3(a0, a1);
    float pb = dot3(d0, d1);
    pa += __shfl_xor(pa, 16, 32);
    pb += __shfl_xor(pb, 16, 32);
    if (lane < 16) {
      brow[ja] = pa + b2s;
      brow[jb] = pb + b2s;
    }
  }
  { // tail: tile 8
    int j = 8 * 16 + p;
    v16h Bh = build_B(j);
    v8f c0, c1;
#pragma unroll
    for (int r = 0; r < 8; ++r) { c0[r] = b1l0[r]; c1[r] = b1l1[r]; }
    c0 = __builtin_amdgcn_wmma_f32_16x16x32_f16(false, W1A0, false, Bh, (short)0, c0, false, false);
    c1 = __builtin_amdgcn_wmma_f32_16x16x32_f16(false, W1A1, false, Bh, (short)0, c1, false, false);
    float part = dot3(c0, c1);
    part += __shfl_xor(part, 16, 32);
    if (lane < 16) brow[j] = part + b2s;
  }
}

// ---------------------------------------------------------------------------
// 7) attention: sim (WMMA, 2 tiles/iter) + bias -> softmax (LDS) -> attn@v
// ---------------------------------------------------------------------------
__global__ void __launch_bounds__(32)
attn_kernel(const _Float16* __restrict__ qT, const _Float16* __restrict__ kB,
            const _Float16* __restrict__ vB, const float* __restrict__ bias,
            float* __restrict__ outb) {
  __shared__ __align__(16) float simS[16 * JP];
  __shared__ __align__(16) _Float16 attnS[16 * JP];
  int bh = blockIdx.x, it = blockIdx.y;
  int i0 = it * 16;
  int lane = threadIdx.x;
  int mrow = lane & 15;
  int c0 = (lane >= 16) ? 8 : 0;
  int mb = (lane >> 4) * 8;
  int n = lane & 15;

  const _Float16* qrow = qT + ((size_t)bh * HW + i0 + mrow) * 32;
  v16h Aq;
  {
    v8h lo = *(const v8h*)(qrow + c0);
    v8h hi = *(const v8h*)(qrow + 16 + c0);
#pragma unroll
    for (int t = 0; t < 8; ++t) { Aq[t] = lo[t]; Aq[8 + t] = hi[t]; }
  }

  for (int jt = 0; jt < 10; jt += 2) { // two independent sim tiles per iter
    v16h Bk0 = *(const v16h*)(kB + ((size_t)(bh * 10 + jt) * 32 + lane) * 16);
    v16h Bk1 = *(const v16h*)(kB + ((size_t)(bh * 10 + jt + 1) * 32 + lane) * 16);
    if (jt < 8)
      __builtin_prefetch(kB + ((size_t)(bh * 10 + jt + 2) * 32 + lane) * 16, 0, 3);
    v8f a0 = {}, a1 = {};
    a0 = __builtin_amdgcn_wmma_f32_16x16x32_f16(false, Aq, false, Bk0, (short)0, a0, false, false);
    a1 = __builtin_amdgcn_wmma_f32_16x16x32_f16(false, Aq, false, Bk1, (short)0, a1, false, false);
#pragma unroll
    for (int u = 0; u < 2; ++u) {
      int j = (jt + u) * 16 + n;
#pragma unroll
      for (int r = 0; r < 8; ++r) {
        int mm = mb + r;
        float v = (u == 0) ? a0[r] : a1[r];
        if (j < Jj) v += bias[((size_t)bh * HW + i0 + mm) * Jj + j];
        else        v = -1e30f;
        simS[mm * JP + (jt + u) * 16 + n] = v;
      }
    }
  }
  __syncthreads();

  if (lane < 16) { // one lane per query row
    float mx = -1e30f;
    for (int j = 0; j < JP; ++j) mx = fmaxf(mx, simS[lane * JP + j]);
    float sum = 0.f;
    for (int j = 0; j < JP; ++j) {
      float e = expf(simS[lane * JP + j] - mx);
      simS[lane * JP + j] = e;
      sum += e;
    }
    float inv = 1.f / sum;
    for (int j = 0; j < JP; ++j) attnS[lane * JP + j] = (_Float16)(simS[lane * JP + j] * inv);
  }
  __syncthreads();

  v8f o0 = {}, o1 = {};
  for (int kt = 0; kt < 5; ++kt) {
    const _Float16* arow = attnS + mrow * JP + kt * 32;
    v8h lo = *(const v8h*)(arow + c0);
    v8h hi = *(const v8h*)(arow + 16 + c0);
    v16h Aa;
#pragma unroll
    for (int t = 0; t < 8; ++t) { Aa[t] = lo[t]; Aa[8 + t] = hi[t]; }
    v16h B0 = *(const v16h*)(vB + ((size_t)((bh * 5 + kt) * 2 + 0) * 32 + lane) * 16);
    v16h B1 = *(const v16h*)(vB + ((size_t)((bh * 5 + kt) * 2 + 1) * 32 + lane) * 16);
    o0 = __builtin_amdgcn_wmma_f32_16x16x32_f16(false, Aa, false, B0, (short)0, o0, false, false);
    o1 = __builtin_amdgcn_wmma_f32_16x16x32_f16(false, Aa, false, B1, (short)0, o1, false, false);
  }
  int b = bh / HEADS, h = bh % HEADS;
#pragma unroll
  for (int r = 0; r < 8; ++r) {
    int mm = mb + r;
    float* orow = outb + ((size_t)b * HW + i0 + mm) * Cc + h * 32;
    orow[n] = o0[r];
    orow[16 + n] = o1[r];
  }
}

// ---------------------------------------------------------------------------
// 8) output projection via fp32 WMMA (16x16x4_f32, K=128 in 32 chunks)
// ---------------------------------------------------------------------------
__global__ void __launch_bounds__(32)
outproj_kernel(const float* __restrict__ x, const float* __restrict__ gamma,
               const float* __restrict__ woB, const float* __restrict__ bo,
               const float* __restrict__ outb, float* __restrict__ dout) {
  int b = blockIdx.x, it = blockIdx.y, nt = blockIdx.z;
  int i0 = it * 16;
  int lane = threadIdx.x;
  int m = lane & 15;
  int koff = (lane >= 16) ? 2 : 0;
  float bov = bo[nt * 16 + (lane & 15)];
  v8f acc;
#pragma unroll
  for (int r = 0; r < 8; ++r) acc[r] = bov;
  const float* arow = outb + ((size_t)b * HW + i0 + m) * Cc;
  const float* bbase = woB + (size_t)nt * 32 * 32 * 2;
  for (int kc = 0; kc < 32; ++kc) {
    v2f Av = *(const v2f*)(arow + kc * 4 + koff);
    v2f Bv = *(const v2f*)(bbase + ((size_t)kc * 32 + lane) * 2);
    acc = __builtin_amdgcn_wmma_f32_16x16x4_f32(false, Av, false, Bv, (short)0, acc, false, false);
  }
  float gm = gamma[0];
  int mb = (lane >> 4) * 8;
  int o = nt * 16 + (lane & 15);
  size_t obase = ((size_t)b * Cc + o) * HW + i0 + mb;
  const float* xr = x + obase;
#pragma unroll
  for (int r = 0; r < 8; ++r) {
    float att = acc[r];
    dout[obase + r] = gm * att + xr[r];
    dout[(size_t)Bq * Cc * HW + obase + r] = att;
  }
}

// ---------------------------------------------------------------------------
extern "C" void kernel_launch(void* const* d_in, const int* in_sizes, int n_in,
                              void* d_out, int out_size, void* d_ws, size_t ws_size,
                              hipStream_t stream) {
  (void)in_sizes; (void)n_in; (void)out_size; (void)ws_size;
  const float* x     = (const float*)d_in[0];
  const float* gamma = (const float*)d_in[1];
  const float* wq    = (const float*)d_in[2];
  const float* wk    = (const float*)d_in[3];
  const float* wv    = (const float*)d_in[4];
  const float* wo    = (const float*)d_in[5];
  const float* bo    = (const float*)d_in[6];
  const float* wdw   = (const float*)d_in[7];
  const float* bdw   = (const float*)d_in[8];
  const float* wproj = (const float*)d_in[9];
  const float* cw0   = (const float*)d_in[10];
  const float* cb0   = (const float*)d_in[11];
  const float* cw1   = (const float*)d_in[12];
  const float* cb1   = (const float*)d_in[13];
  const float* cw2   = (const float*)d_in[14];
  const float* cb2   = (const float*)d_in[15];

  char* base = (char*)d_ws;
  size_t off = 0;
  auto carve = [&](size_t bytes) -> char* {
    char* p = base + off;
    off += (bytes + 255) & ~(size_t)255;
    return p;
  };
  float*    qf   = (float*)carve((size_t)BG * 32 * HW * 4);
  _Float16* qT   = (_Float16*)carve((size_t)BG * HW * 32 * 2);
  float*    vn   = (float*)carve((size_t)BG * 2 * Jj * 4);
  float*    pxy  = (float*)carve((size_t)BG * 2 * Jj * 4);
  float*    kvb  = (float*)carve((size_t)BG * 32 * Jj * 4);
  float*    kf   = (float*)carve((size_t)BG * 32 * Jj * 4);
  float*    vf   = (float*)carve((size_t)BG * 32 * Jj * 4);
  _Float16* kB   = (_Float16*)carve((size_t)BG * 10 * 32 * 16 * 2);
  _Float16* vB   = (_Float16*)carve((size_t)BG * 5 * 2 * 32 * 16 * 2);
  _Float16* w1a  = (_Float16*)carve((size_t)2 * 32 * 16 * 2);
  float*    woB  = (float*)carve((size_t)8 * 32 * 32 * 2 * 4);
  float*    bias = (float*)carve((size_t)BG * HW * Jj * 4);
  float*    outb = (float*)carve((size_t)Bq * HW * Cc * 4);

  q_kernel<<<dim3(BG, HW / 256), 256, 0, stream>>>(x, wq, qf, qT);
  offsets_kernel<<<dim3(BG, Jj), 32, 0, stream>>>(qf, wdw, bdw, wproj, vn, pxy);
  sample_kernel<<<dim3(BG, Jj), 32, 0, stream>>>(x, pxy, kvb);
  kv_kernel<<<(BG * 32 * Jj + 255) / 256, 256, 0, stream>>>(kvb, wk, wv, kf, vf);
  pack_k_kernel<<<BG * 10, 32, 0, stream>>>(kf, kB);
  pack_v_kernel<<<BG * 10, 32, 0, stream>>>(vf, vB);
  prep_w1t_kernel<<<1, 64, 0, stream>>>(cw1, w1a);
  prep_wo_kernel<<<32, 256, 0, stream>>>(wo, woB);
  cpb_kernel<<<dim3(BG, HW), 32, 0, stream>>>(vn, w1a, cw0, cb0, cb1, cw2, cb2, bias);
  attn_kernel<<<dim3(BG, HW / 16), 32, 0, stream>>>(qT, kB, vB, bias, outb);
  outproj_kernel<<<dim3(Bq, HW / 16, Cc / 16), 32, 0, stream>>>(x, gamma, woB, bo, outb, (float*)d_out);
}
